// ResidualBlock_56470230008549
// MI455X (gfx1250) — matrix operand order repaired and checked by reading, not compiled, not run on hardware
//
#include <hip/hip_runtime.h>

#define N_VOX   100000
#define C_CH    64
#define K_OFF   27
#define NSTRIPS (N_VOX / 16)            // 6250 strips of 16 voxels
#define NSUPER  (NSTRIPS / 2)           // 3125 superstrips of 32 voxels (exact)
#define NBLK    ((NSUPER + 7) / 8)      // 391 conv blocks (8 waves each)

typedef __attribute__((ext_vector_type(16))) _Float16 v16h;
typedef __attribute__((ext_vector_type(8)))  _Float16 v8h;
typedef __attribute__((ext_vector_type(8)))  float    v8f;

static __device__ __forceinline__ v8f wmma_f16(v16h a, v16h b, v8f c) {
  return __builtin_amdgcn_wmma_f32_16x16x32_f16(false, a, false, b,
                                                (short)0, c, false, false);
}

// ---------------- prep kernels ----------------

__global__ void cvt_f32_to_f16(const float* __restrict__ src,
                               _Float16* __restrict__ dst, int n) {
  for (int i = blockIdx.x * blockDim.x + threadIdx.x; i < n;
       i += gridDim.x * blockDim.x)
    dst[i] = (_Float16)src[i];
}

// w: [K][C][C] as w[k][c][d]  ->  wt: [K][C][C] as wt[k][d][c] (f16)
__global__ void transpose_w_f16(const float* __restrict__ w,
                                _Float16* __restrict__ wt) {
  int i = blockIdx.x * blockDim.x + threadIdx.x;
  if (i >= K_OFF * C_CH * C_CH) return;
  int k = i / (C_CH * C_CH);
  int r = i - k * C_CH * C_CH;
  int d = r >> 6;
  int c = r & 63;
  wt[i] = (_Float16)w[(k * C_CH + c) * C_CH + d];
}

// ---------------- gather-GEMM via WMMA, fused BN partial stats ----------------
// One wave = 2 strips (32 voxel rows) x 64 output channels.
// A (16x32 f16): lane l holds row m=l&15; 16 consecutive channels at c0+16h, h=l>>4
// B (32x16 f16): lane l holds column d=m; 16 consecutive c at c0+16h -> uses wt[k][d][c]
__global__ __launch_bounds__(256)
void subm_conv_wmma(const _Float16* __restrict__ fin,
                    const int* __restrict__ nbr,
                    const _Float16* __restrict__ wt,
                    float* __restrict__ yout,
                    float* __restrict__ parts) {
  __shared__ float smS[16 * 64];
  __shared__ float smQ[16 * 64];

  const int lane = threadIdx.x & 31;
  const int wv   = threadIdx.x >> 5;
  const int ss   = blockIdx.x * 8 + wv;          // superstrip (wave-uniform)
  const int m    = lane & 15;
  const int h    = lane >> 4;
  const bool active = (ss < NSUPER);             // wave-uniform

  const v16h hzero = {};
  v8f acc[2][4] = {};                            // [strip][column tile]

  if (active) {                                  // EXEC all-ones inside
    const int n0   = ss * 32;
    const int rowA = n0 + m;
    const int rowB = n0 + 16 + m;

    for (int k = 0; k < K_OFF; ++k) {
      const int idx0 = nbr[k * N_VOX + rowA];
      const int idx1 = nbr[k * N_VOX + rowB];
      const _Float16* a0 = fin + (size_t)(idx0 < 0 ? 0 : idx0) * C_CH;
      const _Float16* a1 = fin + (size_t)(idx1 < 0 ? 0 : idx1) * C_CH;
      const _Float16* wk = wt + (size_t)k * C_CH * C_CH + (size_t)m * C_CH + 16 * h;

#pragma unroll
      for (int ch = 0; ch < 2; ++ch) {
        const int c0 = ch * 32;
        union { v16h v; v8h p[2]; } A0, A1;
        A0.p[0] = *(const v8h*)(a0 + c0 + 16 * h);
        A0.p[1] = *(const v8h*)(a0 + c0 + 16 * h + 8);
        if (idx0 < 0) A0.v = hzero;
        A1.p[0] = *(const v8h*)(a1 + c0 + 16 * h);
        A1.p[1] = *(const v8h*)(a1 + c0 + 16 * h + 8);
        if (idx1 < 0) A1.v = hzero;

#pragma unroll
        for (int t = 0; t < 4; ++t) {
          v16h b = *(const v16h*)(wk + c0 + (size_t)(t * 16) * C_CH);
          acc[0][t] = wmma_f16(A0.v, b, acc[0][t]);   // B reused across strips
          acc[1][t] = wmma_f16(A1.v, b, acc[1][t]);
        }
      }
    }

    // C/D layout: VGPR j = element (M = j + 8*h, N = m)
    float* yb = yout + (size_t)n0 * C_CH;
#pragma unroll
    for (int j = 0; j < 8; ++j) {
      const size_t r0 = (size_t)(j + 8 * h) * C_CH;
      const size_t r1 = (size_t)(16 + j + 8 * h) * C_CH;
#pragma unroll
      for (int t = 0; t < 4; ++t) {
        yb[r0 + t * 16 + m] = acc[0][t][j];
        yb[r1 + t * 16 + m] = acc[1][t][j];
      }
    }
  }

  // deterministic per-block BN partials (inactive waves contribute zeros)
#pragma unroll
  for (int t = 0; t < 4; ++t) {
    float s = 0.f, q = 0.f;
#pragma unroll
    for (int i = 0; i < 2; ++i)
#pragma unroll
      for (int j = 0; j < 8; ++j) {
        float v = acc[i][t][j];
        s += v;
        q += v * v;
      }
    smS[(wv * 2 + h) * 64 + t * 16 + m] = s;
    smQ[(wv * 2 + h) * 64 + t * 16 + m] = q;
  }
  __syncthreads();
  if (threadIdx.x < 64) {
    float S = 0.f, Q = 0.f;
#pragma unroll
    for (int w = 0; w < 16; ++w) {
      S += smS[w * 64 + threadIdx.x];
      Q += smQ[w * 64 + threadIdx.x];
    }
    parts[blockIdx.x * 128 + threadIdx.x]      = S;
    parts[blockIdx.x * 128 + 64 + threadIdx.x] = Q;
  }
}

// ---------------- BN finalize (deterministic, fixed-order) ----------------

__global__ void finalize_stats(const float* __restrict__ parts,
                               const float* __restrict__ gamma,
                               const float* __restrict__ beta,
                               float* __restrict__ sshift) {
  const int c = threadIdx.x;               // 64 threads
  float s = 0.f, q = 0.f;
  for (int p = 0; p < NBLK; ++p) {
    s += parts[p * 128 + c];
    q += parts[p * 128 + 64 + c];
  }
  const float invN = 1.0f / (float)N_VOX;
  float mean = s * invN;
  float var  = q * invN - mean * mean;     // biased, as torch BN training
  float rstd = rsqrtf(var + 1e-5f);
  float scale = gamma[c] * rstd;
  sshift[c]      = scale;
  sshift[64 + c] = beta[c] - mean * scale;
}

// ---------------- elementwise epilogues ----------------

__global__ void bn_relu_to_f16(const float* __restrict__ y,
                               const float* __restrict__ ss,
                               _Float16* __restrict__ dst) {
  const int total = N_VOX * C_CH;
  for (int i = blockIdx.x * blockDim.x + threadIdx.x; i < total;
       i += gridDim.x * blockDim.x) {
    int c = i & 63;
    float v = fmaf(y[i], ss[c], ss[64 + c]);
    dst[i] = (_Float16)fmaxf(v, 0.0f);
  }
}

__global__ void bn_add_relu_out(const float* __restrict__ y,
                                const float* __restrict__ ss,
                                const float* __restrict__ x,
                                float* __restrict__ out) {
  const int total = N_VOX * C_CH;
  for (int i = blockIdx.x * blockDim.x + threadIdx.x; i < total;
       i += gridDim.x * blockDim.x) {
    int c = i & 63;
    float v = fmaf(y[i], ss[c], ss[64 + c]) + x[i];
    out[i] = fmaxf(v, 0.0f);
  }
}

// ---------------- workspace layout ----------------

static constexpr size_t align_up(size_t x) { return (x + 255) & ~(size_t)255; }
static constexpr size_t SZ_XH = (size_t)N_VOX * C_CH * 2;           // f16 features
static constexpr size_t SZ_WT = (size_t)K_OFF * C_CH * C_CH * 2;    // f16 weights
static constexpr size_t SZ_Y  = (size_t)N_VOX * C_CH * 4;           // f32 conv out
static constexpr size_t OFF_XH    = 0;           // reused as f1h (xh dead after conv1)
static constexpr size_t OFF_W1T   = align_up(OFF_XH + SZ_XH);
static constexpr size_t OFF_W2T   = align_up(OFF_W1T + SZ_WT);
static constexpr size_t OFF_Y     = align_up(OFF_W2T + SZ_WT);
static constexpr size_t OFF_PARTS = align_up(OFF_Y + SZ_Y);
static constexpr size_t OFF_SS1   = align_up(OFF_PARTS + (size_t)NBLK * 128 * 4);
static constexpr size_t OFF_SS2   = OFF_SS1 + 512;

extern "C" void kernel_launch(void* const* d_in, const int* in_sizes, int n_in,
                              void* d_out, int out_size, void* d_ws, size_t ws_size,
                              hipStream_t stream) {
  const float* x   = (const float*)d_in[0];
  const int*   nbr = (const int*)d_in[1];
  const float* W1  = (const float*)d_in[2];
  const float* g1  = (const float*)d_in[3];
  const float* b1  = (const float*)d_in[4];
  const float* W2  = (const float*)d_in[5];
  const float* g2  = (const float*)d_in[6];
  const float* b2  = (const float*)d_in[7];
  float* out = (float*)d_out;

  char* ws = (char*)d_ws;
  _Float16* xh    = (_Float16*)(ws + OFF_XH);
  _Float16* f1h   = (_Float16*)(ws + OFF_XH);   // alias: xh dead after conv1
  _Float16* w1t   = (_Float16*)(ws + OFF_W1T);
  _Float16* w2t   = (_Float16*)(ws + OFF_W2T);
  float*    ybuf  = (float*)(ws + OFF_Y);
  float*    parts = (float*)(ws + OFF_PARTS);
  float*    ss1   = (float*)(ws + OFF_SS1);
  float*    ss2   = (float*)(ws + OFF_SS2);

  const int wElems = K_OFF * C_CH * C_CH;

  // stage 0: precision conversion / weight transpose
  cvt_f32_to_f16<<<2048, 256, 0, stream>>>(x, xh, N_VOX * C_CH);
  transpose_w_f16<<<(wElems + 255) / 256, 256, 0, stream>>>(W1, w1t);
  transpose_w_f16<<<(wElems + 255) / 256, 256, 0, stream>>>(W2, w2t);

  // stage 1: conv1 (+fused BN partials) -> BN1 finalize -> ReLU (f16 for conv2)
  subm_conv_wmma<<<NBLK, 256, 0, stream>>>(xh, nbr, w1t, ybuf, parts);
  finalize_stats<<<1, 64, 0, stream>>>(parts, g1, b1, ss1);
  bn_relu_to_f16<<<2048, 256, 0, stream>>>(ybuf, ss1, f1h);

  // stage 2: conv2 (+fused BN partials) -> BN2 finalize -> +x -> ReLU
  subm_conv_wmma<<<NBLK, 256, 0, stream>>>(f1h, nbr, w2t, ybuf, parts);
  finalize_stats<<<1, 64, 0, stream>>>(parts, g2, b2, ss2);
  bn_add_relu_out<<<2048, 256, 0, stream>>>(ybuf, ss2, x, out);
}
